// PointerMoHTransformerBlock_8770323219239
// MI455X (gfx1250) — matrix operand order repaired and checked by reading, not compiled, not run on hardware
//
#include <hip/hip_runtime.h>
#include <hip/hip_bf16.h>
#include <math.h>

// ---------------------------------------------------------------------------
// CDNA5 (gfx1250) wave32 WMMA types & helpers
// ---------------------------------------------------------------------------
typedef __bf16 bf16;
typedef __attribute__((ext_vector_type(16))) __bf16 v16bf;
typedef __attribute__((ext_vector_type(8)))  float  v8f;

#define WMMA_BF16(a, b, c) \
    __builtin_amdgcn_wmma_f32_16x16x32_bf16(false, (a), false, (b), (short)0, (c), false, false)

// Async global->LDS copy of 16 bytes per lane (tracked by ASYNCcnt).
__device__ __forceinline__ void async_b128(void* lds_dst, const void* src) {
    unsigned off = (unsigned)(size_t)lds_dst;   // low 32 bits of shared aperture = LDS addr
    asm volatile("global_load_async_to_lds_b128 %0, %1, off"
                 :: "v"(off), "v"(src)
                 : "memory");
}
template <int N> __device__ __forceinline__ void async_wait() {
    asm volatile("s_wait_asynccnt %0" :: "n"(N) : "memory");
}
// compiler-only fence: keep LDS store->load order within a wave (LDS is
// in-order per wave in hardware)
__device__ __forceinline__ void lds_fence() {
    asm volatile("" ::: "memory");
}

// A-matrix fragment (16x32 bf16) from row-major LDS tile, per ISA layout:
// lane(hi,row): contiguous 8-elem segments at K = grp*16 + hi*8   (grp=0,1)
__device__ __forceinline__ v16bf load_a_frag(const bf16* t, int lda, int lane) {
    int hi  = lane >> 4;
    const bf16* p = t + (lane & 15) * lda + (hi << 3);
    v16bf f;
#pragma unroll
    for (int e = 0; e < 8; ++e) f[e] = p[e];
#pragma unroll
    for (int e = 0; e < 8; ++e) f[8 + e] = p[16 + e];
    return f;
}

// B-matrix fragment (32x16) where LDS tile holds B^T row-major: t[n][k].
// lane(hi,col): contiguous 8-elem segments at row=col, K = grp*16 + hi*8 (grp=0,1).
__device__ __forceinline__ v16bf load_b_seg(const bf16* t, int ld, int lane) {
    const bf16* p = t + (lane & 15) * ld + ((lane >> 4) << 3);
    v16bf f;
#pragma unroll
    for (int e = 0; e < 8; ++e) f[e] = p[e];
#pragma unroll
    for (int e = 0; e < 8; ++e) f[8 + e] = p[16 + e];
    return f;
}

__device__ __forceinline__ float gelu_exact(float x) {
    return 0.5f * x * (1.0f + erff(x * 0.70710678118654752f));
}
__device__ __forceinline__ float hmax16(float x) {
    x = fmaxf(x, __shfl_xor(x, 1, 32));
    x = fmaxf(x, __shfl_xor(x, 2, 32));
    x = fmaxf(x, __shfl_xor(x, 4, 32));
    x = fmaxf(x, __shfl_xor(x, 8, 32));
    return x;
}
__device__ __forceinline__ float hsum16(float x) {
    x += __shfl_xor(x, 1, 32);
    x += __shfl_xor(x, 2, 32);
    x += __shfl_xor(x, 4, 32);
    x += __shfl_xor(x, 8, 32);
    return x;
}

// ---------------------------------------------------------------------------
// Weight prep: Wt[n][kp] = (kp<K) ? bf16(W[kp][n]) : 0   (row stride Kpad)
// ---------------------------------------------------------------------------
__global__ __launch_bounds__(256) void wtrans_kernel(
    const float* __restrict__ W, bf16* __restrict__ Wt, int K, int N, int Kpad) {
    __shared__ float t[32][33];
    int k0 = blockIdx.y * 32, n0 = blockIdx.x * 32;
    int r = threadIdx.x >> 5, c = threadIdx.x & 31;
#pragma unroll
    for (int i = 0; i < 4; ++i) {
        int k = k0 + r + 8 * i;
        t[r + 8 * i][c] = (k < K) ? W[(size_t)k * N + n0 + c] : 0.0f;
    }
    __syncthreads();
#pragma unroll
    for (int i = 0; i < 4; ++i) {
        int n = n0 + r + 8 * i;
        Wt[(size_t)n * Kpad + k0 + c] = (bf16)t[c][r + 8 * i];
    }
}

// ---------------------------------------------------------------------------
// WMMA GEMM:  out = epi(A[M,K]bf16 @ Wt[N,K]bf16^T + bias [+resid])
// 256 thr (8 waves), BM=128 BN=128 BK=64, waves 4x2 -> 32x64 each,
// 16 WMMA per wave per K-tile. Async double-buffered staging.
// K % 64 == 0, M % 128 == 0, N % 128 == 0.
// mode 0: Cf fp32 (+resid) ; mode 1: Ch bf16 row-major ; mode 2: Ch bf16
// transposed as [b][col][t] with T=2048,D=1024 (for V).
// ---------------------------------------------------------------------------
__global__ __launch_bounds__(256) void gemm_kernel(
    const bf16* __restrict__ A, const bf16* __restrict__ Wt,
    const float* __restrict__ bias, const float* __restrict__ resid,
    float* __restrict__ Cf, bf16* __restrict__ Ch,
    int M, int N, int K, int act, int mode) {
    const int LDA = 72, LDB = 72;
    __shared__ bf16 As[2][128][LDA];
    __shared__ bf16 Bs[2][128][LDB];

    int tid  = threadIdx.x;
    int lane = tid & 31;
    int wave = tid >> 5;
    int wm   = wave >> 1, wn = wave & 1;
    int bm   = blockIdx.y * 128;
    int bn   = blockIdx.x * 128;

    auto issue = [&](int buf, int k0) {
#pragma unroll
        for (int i = 0; i < 4; ++i) {
            int id  = tid + i * 256;          // 0..1023
            int row = id >> 3, cq = (id & 7) * 8;
            async_b128(&As[buf][row][cq], A  + (size_t)(bm + row) * K + k0 + cq);
            async_b128(&Bs[buf][row][cq], Wt + (size_t)(bn + row) * K + k0 + cq);
        }
    };

    v8f acc[2][4] = {};
    auto compute = [&](int buf) {
#pragma unroll
        for (int kk = 0; kk < 2; ++kk) {
            v16bf a0 = load_a_frag(&As[buf][wm * 32][kk * 32],      LDA, lane);
            v16bf a1 = load_a_frag(&As[buf][wm * 32 + 16][kk * 32], LDA, lane);
#pragma unroll
            for (int j = 0; j < 4; ++j) {
                v16bf bj = load_b_seg(&Bs[buf][wn * 64 + j * 16][kk * 32], LDB, lane);
                acc[0][j] = WMMA_BF16(a0, bj, acc[0][j]);
                acc[1][j] = WMMA_BF16(a1, bj, acc[1][j]);
            }
        }
    };

    int nt = K >> 6;
    issue(0, 0);
    for (int t = 0; t < nt - 1; ++t) {
        issue((t + 1) & 1, (t + 1) << 6);
        async_wait<8>();
        __syncthreads();
        compute(t & 1);
        __syncthreads();
    }
    async_wait<0>();
    __syncthreads();
    compute((nt - 1) & 1);

    int hi = lane >> 4, lo = lane & 15;
#pragma unroll
    for (int i = 0; i < 2; ++i) {
#pragma unroll
        for (int j = 0; j < 4; ++j) {
#pragma unroll
            for (int rr = 0; rr < 8; ++rr) {
                int row = bm + wm * 32 + i * 16 + rr + (hi << 3);
                int col = bn + wn * 64 + j * 16 + lo;
                float v = acc[i][j][rr] + bias[col];
                if (act) v = gelu_exact(v);
                if (mode == 0) {
                    size_t idx = (size_t)row * N + col;
                    if (resid) v += resid[idx];
                    Cf[idx] = v;
                } else if (mode == 1) {
                    Ch[(size_t)row * N + col] = (bf16)v;
                } else {  // transposed V: [b][col][t]
                    int bb = row >> 11, tr = row & 2047;
                    Ch[((size_t)(bb * 1024 + col)) * 2048 + tr] = (bf16)v;
                }
            }
        }
    }
}

// ---------------------------------------------------------------------------
// Flash attention with WMMA. Q,K bf16 [B*T, D] (head at col h*64);
// V bf16 transposed [B][D][T]. O fp32 [B*T, D]. 128 thr (4 waves),
// 64 queries/block; async double-buffered K/V tiles.
// ---------------------------------------------------------------------------
__global__ __launch_bounds__(128) void attn_kernel(
    const bf16* __restrict__ Qg, const bf16* __restrict__ Kg,
    const bf16* __restrict__ Vt, float* __restrict__ O) {
    const int T = 2048, D = 1024, LD = 72;
    __shared__ bf16 Qs[64][LD];
    __shared__ bf16 Ks[2][64][LD];
    __shared__ bf16 Vs[2][64][LD];   // holds V^T tile: [dh][key]
    __shared__ bf16 Ps[4][16][LD];

    int tid  = threadIdx.x;
    int lane = tid & 31;
    int w    = tid >> 5;
    int hi   = lane >> 4, lo = lane & 15;

    int bh = blockIdx.y;
    int b  = bh >> 4;
    int h  = bh & 15;
    int q0 = blockIdx.x * 64;

    auto issueKV = [&](int buf, int kt) {
#pragma unroll
        for (int i = 0; i < 4; ++i) {
            int id = tid + i * 128;         // 0..511
            int row = id >> 3, cq = (id & 7) * 8;
            async_b128(&Ks[buf][row][cq],
                       Kg + ((size_t)(b * T + kt * 64 + row)) * D + h * 64 + cq);
            async_b128(&Vs[buf][row][cq],
                       Vt + ((size_t)(b * D + h * 64 + row)) * T + kt * 64 + cq);
        }
    };

    // stage Q tile (async), then first K/V tile
#pragma unroll
    for (int i = 0; i < 4; ++i) {
        int id = tid + i * 128;
        int row = id >> 3, cq = (id & 7) * 8;
        async_b128(&Qs[row][cq],
                   Qg + ((size_t)(b * T + q0 + row)) * D + h * 64 + cq);
    }
    issueKV(0, 0);

    v8f   o[4] = {};
    float m[8], l[8];
#pragma unroll
    for (int rr = 0; rr < 8; ++rr) { m[rr] = -INFINITY; l[rr] = 0.0f; }

    for (int kt = 0; kt < T / 64; ++kt) {
        int buf = kt & 1;
        if (kt + 1 < T / 64) { issueKV(buf ^ 1, kt + 1); async_wait<8>(); }
        else                 { async_wait<0>(); }
        __syncthreads();

        // S = Q K^T  (16 rows x 64 keys per wave)
        v16bf aq0 = load_a_frag(&Qs[w * 16][0],  LD, lane);
        v16bf aq1 = load_a_frag(&Qs[w * 16][32], LD, lane);
        v8f s[4] = {};
#pragma unroll
        for (int j = 0; j < 4; ++j) {
            // logical B[k=dh][n=key] = Ks[key][dh] -> row segment of Ks
            v16bf bk0 = load_b_seg(&Ks[buf][j * 16][0],  LD, lane);
            s[j] = WMMA_BF16(aq0, bk0, s[j]);
            v16bf bk1 = load_b_seg(&Ks[buf][j * 16][32], LD, lane);
            s[j] = WMMA_BF16(aq1, bk1, s[j]);
        }

        // online softmax
#pragma unroll
        for (int rr = 0; rr < 8; ++rr) {
            float pv[4];
            float mx = -INFINITY;
#pragma unroll
            for (int j = 0; j < 4; ++j) {
                float x = s[j][rr] * 0.125f;
                pv[j] = x;
                mx = fmaxf(mx, x);
            }
            mx = hmax16(mx);
            float mn   = fmaxf(m[rr], mx);
            float corr = expf(m[rr] - mn);
            m[rr] = mn;
            float ls = 0.0f;
#pragma unroll
            for (int j = 0; j < 4; ++j) {
                float p = expf(pv[j] - mn);
                ls += p;
                Ps[w][rr + (hi << 3)][j * 16 + lo] = (bf16)p;
            }
            ls = hsum16(ls);
            l[rr] = l[rr] * corr + ls;
#pragma unroll
            for (int j = 0; j < 4; ++j) o[j][rr] *= corr;
        }
        // same-wave LDS store->load ordering only (each wave reads its own Ps
        // region; LDS is in-order per wave) -> compiler fence suffices
        lds_fence();

        // O += P V : logical B[k=key][n=dh] = Vs[dh][key] -> row segment
#pragma unroll
        for (int c = 0; c < 2; ++c) {
            v16bf ap = load_a_frag(&Ps[w][0][c * 32], LD, lane);
#pragma unroll
            for (int j = 0; j < 4; ++j) {
                v16bf bv = load_b_seg(&Vs[buf][j * 16][c * 32], LD, lane);
                o[j] = WMMA_BF16(ap, bv, o[j]);
            }
        }
        __syncthreads();
    }

#pragma unroll
    for (int j = 0; j < 4; ++j) {
#pragma unroll
        for (int rr = 0; rr < 8; ++rr) {
            int row = q0 + w * 16 + rr + (hi << 3);
            int col = h * 64 + j * 16 + lo;
            O[((size_t)(b * T + row)) * D + col] = o[j][rr] / l[rr];
        }
    }
}

// ---------------------------------------------------------------------------
// LayerNorm over D=1024, one block per row; input fp32 or bf16, output bf16
// ---------------------------------------------------------------------------
template <typename TI>
__global__ __launch_bounds__(256) void ln_kernel(
    const TI* __restrict__ x, const float* __restrict__ g,
    const float* __restrict__ bb, bf16* __restrict__ y) {
    const int D = 1024;
    int row = blockIdx.x;
    const TI* xr = x + (size_t)row * D + threadIdx.x * 4;
    float v0 = (float)xr[0], v1 = (float)xr[1], v2 = (float)xr[2], v3 = (float)xr[3];
    float s = v0 + v1 + v2 + v3;
    float q = v0 * v0 + v1 * v1 + v2 * v2 + v3 * v3;
#pragma unroll
    for (int msk = 1; msk < 32; msk <<= 1) {
        s += __shfl_xor(s, msk, 32);
        q += __shfl_xor(q, msk, 32);
    }
    __shared__ float rs[8], rq[8];
    int wave = threadIdx.x >> 5, lane = threadIdx.x & 31;
    if (lane == 0) { rs[wave] = s; rq[wave] = q; }
    __syncthreads();
    if (threadIdx.x == 0) {
        float ts = 0, tq = 0;
#pragma unroll
        for (int i = 0; i < 8; ++i) { ts += rs[i]; tq += rq[i]; }
        rs[0] = ts; rq[0] = tq;
    }
    __syncthreads();
    float mean = rs[0] * (1.0f / D);
    float var  = rq[0] * (1.0f / D) - mean * mean;
    float rstd = rsqrtf(var + 1e-5f);
    int c = threadIdx.x * 4;
    const float* gp = g + c;
    const float* bp = bb + c;
    bf16* yp = y + (size_t)row * D + c;
    yp[0] = (bf16)((v0 - mean) * rstd * gp[0] + bp[0]);
    yp[1] = (bf16)((v1 - mean) * rstd * gp[1] + bp[1]);
    yp[2] = (bf16)((v2 - mean) * rstd * gp[2] + bp[2]);
    yp[3] = (bf16)((v3 - mean) * rstd * gp[3] + bp[3]);
}

// ---------------------------------------------------------------------------
// Tiny matmul C[M,16] = A[M,K]bf16 @ W[K,16]f32 + bias (optional +=)
// ---------------------------------------------------------------------------
__global__ __launch_bounds__(256) void small_mm_kernel(
    const bf16* __restrict__ A, const float* __restrict__ W,
    const float* __restrict__ bias, float* __restrict__ C, int K, int addInto) {
    int ty = threadIdx.x >> 4, tx = threadIdx.x & 15;
    int row = blockIdx.x * 16 + ty;
    __shared__ float As[16][17];
    float acc = 0.0f;
    for (int k0 = 0; k0 < K; k0 += 16) {
        As[ty][tx] = (float)A[(size_t)row * K + k0 + tx];
        __syncthreads();
#pragma unroll
        for (int kk = 0; kk < 16; ++kk)
            acc += As[ty][kk] * W[(size_t)(k0 + kk) * 16 + tx];
        __syncthreads();
    }
    acc += bias[tx];
    size_t idx = (size_t)row * 16 + tx;
    if (addInto) C[idx] += acc; else C[idx] = acc;
}

// cat[row] = [ hc(1024) | mean_dh(heads)(16) | zeros(48) ]  stride 1088, bf16
__global__ __launch_bounds__(256) void build_cat_kernel(
    const bf16* __restrict__ hc, const float* __restrict__ heads,
    bf16* __restrict__ cat) {
    int row = blockIdx.x;
    const bf16* sp = hc + (size_t)row * 1024 + threadIdx.x * 4;
    bf16* dp = cat + (size_t)row * 1088 + threadIdx.x * 4;
    dp[0] = sp[0]; dp[1] = sp[1]; dp[2] = sp[2]; dp[3] = sp[3];
    if (threadIdx.x < 64) {
        bf16 v = (bf16)0.0f;
        if (threadIdx.x < 16) {
            const float* hr = heads + (size_t)row * 1024 + threadIdx.x * 64;
            float sm = 0.0f;
#pragma unroll
            for (int i = 0; i < 64; ++i) sm += hr[i];
            v = (bf16)(sm * (1.0f / 64.0f));
        }
        cat[(size_t)row * 1088 + 1024 + threadIdx.x] = v;
    }
}

// alphas = softmax(logits/tau); heads_b = bf16(heads * alpha)
__global__ __launch_bounds__(256) void route_scale_kernel(
    const float* __restrict__ logits, const float* __restrict__ heads,
    bf16* __restrict__ heads_b) {
    int row = blockIdx.x;
    __shared__ float lg[16];
    if (threadIdx.x < 16)
        lg[threadIdx.x] = logits[(size_t)row * 16 + threadIdx.x] * (1.0f / 0.7f);
    __syncthreads();
    float mx = -INFINITY;
#pragma unroll
    for (int hh = 0; hh < 16; ++hh) mx = fmaxf(mx, lg[hh]);
    float e[16];
    float sm = 0.0f;
#pragma unroll
    for (int hh = 0; hh < 16; ++hh) { e[hh] = expf(lg[hh] - mx); sm += e[hh]; }
    float inv = 1.0f / sm;
    int c0 = threadIdx.x * 4;
#pragma unroll
    for (int i = 0; i < 4; ++i) {
        int c = c0 + i;
        heads_b[(size_t)row * 1024 + c] = (bf16)(heads[(size_t)row * 1024 + c] * e[c >> 6] * inv);
    }
}

// attn_out = ctx(bf16) + x(f32), fp32 out
__global__ __launch_bounds__(256) void add_kernel(
    const bf16* __restrict__ a, const float* __restrict__ b,
    float* __restrict__ c, int n) {
    int i = (blockIdx.x * blockDim.x + threadIdx.x) * 4;
    if (i < n) {
#pragma unroll
        for (int k = 0; k < 4; ++k) c[i + k] = (float)a[i + k] + b[i + k];
    }
}

// ---------------------------------------------------------------------------
// host-side orchestration
// ---------------------------------------------------------------------------
extern "C" void kernel_launch(void* const* d_in, const int* in_sizes, int n_in,
                              void* d_out, int out_size, void* d_ws, size_t ws_size,
                              hipStream_t stream) {
    const int T = 2048, D = 1024, BT = 4096, HID = 512, DFF = 4096;
    const int KCAT = 1088;   // 1040 padded to a multiple of 64
    (void)in_sizes; (void)n_in; (void)out_size; (void)ws_size;

    const float* x   = (const float*)d_in[0];
    const float* Wq  = (const float*)d_in[1];  const float* bq  = (const float*)d_in[2];
    const float* Wk  = (const float*)d_in[3];  const float* bk  = (const float*)d_in[4];
    const float* Wv  = (const float*)d_in[5];  const float* bv  = (const float*)d_in[6];
    const float* Wo  = (const float*)d_in[7];  const float* bo  = (const float*)d_in[8];
    const float* g1  = (const float*)d_in[9];  const float* b1  = (const float*)d_in[10];
    const float* g2  = (const float*)d_in[11]; const float* b2  = (const float*)d_in[12];
    const float* gc  = (const float*)d_in[13]; const float* bc  = (const float*)d_in[14];
    const float* Ws1 = (const float*)d_in[15]; const float* bs1 = (const float*)d_in[16];
    const float* Ws2 = (const float*)d_in[17]; const float* bs2 = (const float*)d_in[18];
    const float* Wr1 = (const float*)d_in[19]; const float* br1 = (const float*)d_in[20];
    const float* Wr2 = (const float*)d_in[21]; const float* br2 = (const float*)d_in[22];
    const float* Wf1 = (const float*)d_in[23]; const float* bf1 = (const float*)d_in[24];
    const float* Wf2 = (const float*)d_in[25]; const float* bf2 = (const float*)d_in[26];
    float* out = (float*)d_out;

    char* base = (char*)d_ws;
    size_t off = 0;
    auto alloc = [&](size_t bytes) {
        void* p = base + off;
        off = (off + bytes + 255) & ~(size_t)255;
        return p;
    };
    // activations (order matters: kb..cat form the ffh overlay region)
    bf16* ctx     = (bf16*)alloc((size_t)BT * D * 2);
    bf16* qb      = (bf16*)alloc((size_t)BT * D * 2);
    bf16* kb      = (bf16*)alloc((size_t)BT * D * 2);
    bf16* vT      = (bf16*)alloc((size_t)BT * D * 2);
    bf16* hc      = (bf16*)alloc((size_t)BT * D * 2);
    bf16* hid     = (bf16*)alloc((size_t)BT * HID * 2);
    bf16* cat     = (bf16*)alloc((size_t)BT * KCAT * 2);
    bf16* heads_b = (bf16*)alloc((size_t)BT * D * 2);
    float* heads  = (float*)alloc((size_t)BT * D * 4);
    float* logits = (float*)alloc((size_t)BT * 16 * 4);
    float* attn_o = (float*)alloc((size_t)BT * D * 4);
    // transposed bf16 weights
    bf16* Wqt  = (bf16*)alloc((size_t)D * D * 2);
    bf16* Wkt  = (bf16*)alloc((size_t)D * D * 2);
    bf16* Wvt  = (bf16*)alloc((size_t)D * D * 2);
    bf16* Wot  = (bf16*)alloc((size_t)D * D * 2);
    bf16* Ws1t = (bf16*)alloc((size_t)HID * D * 2);
    bf16* Wr1t = (bf16*)alloc((size_t)HID * KCAT * 2);
    bf16* Wf1t = (bf16*)alloc((size_t)DFF * D * 2);
    bf16* Wf2t = (bf16*)alloc((size_t)D * DFF * 2);
    // overlays (dead regions at FFN time)
    bf16* yin = qb;               // 8MB
    bf16* ffh = kb;               // 32MB spans kb+vT+hc+hid+cat (~37MB)

    dim3 blk(256);
    // ---- weight prep (transpose + bf16 convert) ----
    wtrans_kernel<<<dim3(D / 32, D / 32), blk, 0, stream>>>(Wq, Wqt, D, D, D);
    wtrans_kernel<<<dim3(D / 32, D / 32), blk, 0, stream>>>(Wk, Wkt, D, D, D);
    wtrans_kernel<<<dim3(D / 32, D / 32), blk, 0, stream>>>(Wv, Wvt, D, D, D);
    wtrans_kernel<<<dim3(D / 32, D / 32), blk, 0, stream>>>(Wo, Wot, D, D, D);
    wtrans_kernel<<<dim3(HID / 32, D / 32), blk, 0, stream>>>(Ws1, Ws1t, D, HID, D);
    wtrans_kernel<<<dim3(HID / 32, KCAT / 32), blk, 0, stream>>>(Wr1, Wr1t, D + 16, HID, KCAT);
    wtrans_kernel<<<dim3(DFF / 32, D / 32), blk, 0, stream>>>(Wf1, Wf1t, D, DFF, D);
    wtrans_kernel<<<dim3(D / 32, DFF / 32), blk, 0, stream>>>(Wf2, Wf2t, DFF, D, DFF);

    dim3 gD(D / 128, BT / 128);
    dim3 gH(HID / 128, BT / 128);
    dim3 gF(DFF / 128, BT / 128);
    dim3 gAttn(T / 64, 2 * 16);

    ln_kernel<float><<<BT, blk, 0, stream>>>(x, g1, b1, ctx);

    for (int it = 0; it < 2; ++it) {
        gemm_kernel<<<gD, blk, 0, stream>>>(ctx, Wqt, bq, nullptr, nullptr, qb, BT, D, D, 0, 1);
        gemm_kernel<<<gD, blk, 0, stream>>>(ctx, Wkt, bk, nullptr, nullptr, kb, BT, D, D, 0, 1);
        gemm_kernel<<<gD, blk, 0, stream>>>(ctx, Wvt, bv, nullptr, nullptr, vT, BT, D, D, 0, 2);
        attn_kernel<<<gAttn, dim3(128), 0, stream>>>(qb, kb, vT, heads);
        ln_kernel<bf16><<<BT, blk, 0, stream>>>(ctx, gc, bc, hc);
        gemm_kernel<<<gH, blk, 0, stream>>>(hc, Ws1t, bs1, nullptr, nullptr, hid, BT, HID, D, 1, 1);
        small_mm_kernel<<<BT / 16, blk, 0, stream>>>(hid, Ws2, bs2, logits, HID, 0);
        build_cat_kernel<<<BT, blk, 0, stream>>>(hc, heads, cat);
        gemm_kernel<<<gH, blk, 0, stream>>>(cat, Wr1t, br1, nullptr, nullptr, hid, BT, HID, KCAT, 1, 1);
        small_mm_kernel<<<BT / 16, blk, 0, stream>>>(hid, Wr2, br2, logits, HID, 1);
        route_scale_kernel<<<BT, blk, 0, stream>>>(logits, heads, heads_b);
        gemm_kernel<<<gD, blk, 0, stream>>>(heads_b, Wot, bo, nullptr, nullptr, ctx, BT, D, D, 0, 1);
    }

    add_kernel<<<(BT * D / 4 + 255) / 256, blk, 0, stream>>>(ctx, x, attn_o, BT * D);
    ln_kernel<float><<<BT, blk, 0, stream>>>(attn_o, g2, b2, yin);
    gemm_kernel<<<gF, blk, 0, stream>>>(yin, Wf1t, bf1, nullptr, nullptr, ffh, BT, DFF, D, 1, 1);
    gemm_kernel<<<gD, blk, 0, stream>>>(ffh, Wf2t, bf2, attn_o, out, nullptr, BT, D, DFF, 0, 0);
}
